// ResEG_GRU_687194767632
// MI455X (gfx1250) — compile-verified, matrix operand order checked by reading, not run on hardware
//
#include <hip/hip_runtime.h>
#include <hip/hip_bf16.h>
#include <math.h>

// ---------------- problem constants ----------------
#define BB   2048
#define SS   5
#define NN   80      // nodes
#define EE   592     // edges
#define FX   8
#define FF   4
#define FE   8
#define DD   8
#define HH   800
#define GIN  (NN*DD + EE*DD)   // 5376 GRU input
#define G3H  (3*HH)            // 2400
#define MLP1 1280
#define OUTF (SS*80)           // 400
#define YW   (SS*HH)           // 4000

// ---------------- WMMA types ----------------
typedef __attribute__((ext_vector_type(16))) __bf16 v16bf;
typedef __attribute__((ext_vector_type(8)))  float  v8f;

#define WMMA_BF16(a, b, c) \
  __builtin_amdgcn_wmma_f32_16x16x32_bf16(false, (a), false, (b), (short)0, (c), false, false)

__device__ __forceinline__ unsigned short f2bf(float f) {
  union { float f; unsigned u; } v; v.f = f;
  unsigned u = v.u;
  unsigned r = ((u >> 16) & 1u) + 0x7FFFu;   // round-to-nearest-even
  return (unsigned short)((u + r) >> 16);
}

struct Frag { union { v16bf v; uint4 q[2]; }; };

// Per-lane gather matching the CDNA5 16-bit A/B 16x32 VGPR striping:
// lanes 0-15 hold row (lane), K = {kb..kb+7, kb+16..kb+23}
// lanes 16-31 hold row (lane-16), K = {kb+8..kb+15, kb+24..kb+31}
__device__ __forceinline__ Frag load_frag(const unsigned short* __restrict__ p,
                                          int row_base, int K, int kb, int lane) {
  Frag f;
  const unsigned short* rp = p + (size_t)(row_base + (lane & 15)) * K + kb
                               + ((lane < 16) ? 0 : 8);
  f.q[0] = *reinterpret_cast<const uint4*>(rp);
  f.q[1] = *reinterpret_cast<const uint4*>(rp + 16);
  return f;
}

__device__ __forceinline__ void store_tile(const v8f& acc, float* __restrict__ C,
                                           unsigned short* __restrict__ Cbf,
                                           const float* __restrict__ bias,
                                           int rb, int cb, int ldc, int lane) {
  const int c    = cb + (lane & 15);
  const int roff = (lane >= 16) ? 8 : 0;
  const float bv = bias ? bias[c] : 0.f;
#pragma unroll
  for (int v = 0; v < 8; ++v) {
    float val = acc[v] + bv;
    size_t o = (size_t)(rb + v + roff) * ldc + c;
    if (C)   C[o]   = val;
    if (Cbf) Cbf[o] = f2bf(val);
  }
}

// ---------------- generic bf16 WMMA GEMM:  D[M,N] = A[M,K] * W[N,K]^T + bias ----------------
// 256 threads = 8 waves as 4(M) x 2(N); each wave owns a 64x32 register tile
// (4 A-frags x 2 B-frags -> 8 WMMAs per 32-K step, ~21 FLOP/byte of frag traffic).
// Requires M % 256 == 0, K % 32 == 0, N % 16 == 0. Column handling is hoisted out of
// the k-loop: full 32-col path vs single-16-col edge path (wave-uniform selection,
// EXEC stays all-ones around every WMMA).
__global__ void __launch_bounds__(256)
gemm_bf16_kernel(const unsigned short* __restrict__ A,
                 const unsigned short* __restrict__ W,
                 const float* __restrict__ bias,
                 float* __restrict__ C,
                 unsigned short* __restrict__ Cbf,
                 int M, int N, int K) {
  const int lane = threadIdx.x & 31;
  const int wave = threadIdx.x >> 5;
  const int rb = blockIdx.x * 256 + (wave & 3) * 64;   // rows rb .. rb+63
  const int cb = blockIdx.y * 64  + (wave >> 2) * 32;  // cols cb .. cb+31
  if (cb >= N) return;                                  // wave-uniform

  v8f a00 = {}, a01 = {}, a10 = {}, a11 = {}, a20 = {}, a21 = {}, a30 = {}, a31 = {};

  if (cb + 16 < N) {
    // ---- full path: 64 x 32 ----
    for (int kb = 0; kb < K; kb += 32) {
      if (kb + 64 < K)
        __builtin_prefetch(A + (size_t)(rb + (lane & 15)) * K + kb + 64, 0, 3);
      Frag f0 = load_frag(A, rb,      K, kb, lane);
      Frag f1 = load_frag(A, rb + 16, K, kb, lane);
      Frag f2 = load_frag(A, rb + 32, K, kb, lane);
      Frag f3 = load_frag(A, rb + 48, K, kb, lane);
      Frag b0 = load_frag(W, cb,      K, kb, lane);
      Frag b1 = load_frag(W, cb + 16, K, kb, lane);
      a00 = WMMA_BF16(f0.v, b0.v, a00);
      a10 = WMMA_BF16(f1.v, b0.v, a10);
      a20 = WMMA_BF16(f2.v, b0.v, a20);
      a30 = WMMA_BF16(f3.v, b0.v, a30);
      a01 = WMMA_BF16(f0.v, b1.v, a01);
      a11 = WMMA_BF16(f1.v, b1.v, a11);
      a21 = WMMA_BF16(f2.v, b1.v, a21);
      a31 = WMMA_BF16(f3.v, b1.v, a31);
    }
    store_tile(a00, C, Cbf, bias, rb,      cb,      N, lane);
    store_tile(a10, C, Cbf, bias, rb + 16, cb,      N, lane);
    store_tile(a20, C, Cbf, bias, rb + 32, cb,      N, lane);
    store_tile(a30, C, Cbf, bias, rb + 48, cb,      N, lane);
    store_tile(a01, C, Cbf, bias, rb,      cb + 16, N, lane);
    store_tile(a11, C, Cbf, bias, rb + 16, cb + 16, N, lane);
    store_tile(a21, C, Cbf, bias, rb + 32, cb + 16, N, lane);
    store_tile(a31, C, Cbf, bias, rb + 48, cb + 16, N, lane);
  } else {
    // ---- edge path: 64 x 16 (cb == N-16) ----
    for (int kb = 0; kb < K; kb += 32) {
      Frag f0 = load_frag(A, rb,      K, kb, lane);
      Frag f1 = load_frag(A, rb + 16, K, kb, lane);
      Frag f2 = load_frag(A, rb + 32, K, kb, lane);
      Frag f3 = load_frag(A, rb + 48, K, kb, lane);
      Frag b0 = load_frag(W, cb,      K, kb, lane);
      a00 = WMMA_BF16(f0.v, b0.v, a00);
      a10 = WMMA_BF16(f1.v, b0.v, a10);
      a20 = WMMA_BF16(f2.v, b0.v, a20);
      a30 = WMMA_BF16(f3.v, b0.v, a30);
    }
    store_tile(a00, C, Cbf, bias, rb,      cb, N, lane);
    store_tile(a10, C, Cbf, bias, rb + 16, cb, N, lane);
    store_tile(a20, C, Cbf, bias, rb + 32, cb, N, lane);
    store_tile(a30, C, Cbf, bias, rb + 48, cb, N, lane);
  }
}

// ---------------- weight conversion ----------------
__global__ void cvt_kernel(const float* __restrict__ s, unsigned short* __restrict__ d, int n) {
  int i = blockIdx.x * 256 + threadIdx.x;
  if (i < n) d[i] = f2bf(s[i]);
}
// src [rows, cols] -> dst [cols, rows] (bf16)
__global__ void cvt_t_kernel(const float* __restrict__ s, unsigned short* __restrict__ d,
                             int rows, int cols) {
  int i = blockIdx.x * 256 + threadIdx.x;
  if (i < rows * cols) {
    int r = i / cols, c = i % cols;
    d[(size_t)c * rows + r] = f2bf(s[i]);
  }
}

// ---------------- GNN: one block per graph (B*S graphs) ----------------
__device__ __forceinline__ int wrap80(int i) { i %= NN; return (i < 0) ? i + NN : i; }

__global__ void __launch_bounds__(256)
gnn_kernel(const float* __restrict__ x, const int* __restrict__ edge_index,
           const float* __restrict__ edge_f, const float* __restrict__ edge_attr,
           const float* __restrict__ Wx, const float* __restrict__ We,
           const float* __restrict__ Wf, const float* __restrict__ Wedge,
           unsigned short* __restrict__ seq_bf) {
  const int g   = blockIdx.x;       // 0 .. B*S-1
  const int tid = threadIdx.x;

  __shared__ float s_xout[NN * DD];
  __shared__ float s_agg [NN * DD];
  __shared__ float s_Wx[FX * DD], s_We[FE * DD], s_Wf[FF * DD], s_Wedge[(2 * DD + FE) * DD];
  __shared__ int   s_min;

  if (tid < FX * DD) s_Wx[tid] = Wx[tid];
  if (tid < FE * DD) s_We[tid] = We[tid];
  if (tid < FF * DD) s_Wf[tid] = Wf[tid];
  if (tid < (2 * DD + FE) * DD) s_Wedge[tid] = Wedge[tid];
  if (tid == 0) s_min = 0x7fffffff;
  for (int i = tid; i < NN * DD; i += 256) s_agg[i] = 0.f;
  __syncthreads();

  const int* ei = edge_index + (size_t)g * 2 * EE;
  int lmin = 0x7fffffff;
  for (int e = tid; e < EE; e += 256) lmin = min(lmin, ei[e]);
  atomicMin(&s_min, lmin);
  __syncthreads();
  const int mn = s_min;

  const float* eag = edge_attr + (size_t)g * EE * FE;
  const float* efg = edge_f    + (size_t)g * EE * FF;

  // edge messages + scatter-add to nodes (LDS float atomics)
  for (int e = tid; e < EE; e += 256) {
    const float* ea = eag + e * FE;
    const float* ef = efg + e * FF;
    const int dst = wrap80(ei[EE + e] - mn);
#pragma unroll
    for (int d = 0; d < DD; ++d) {
      float acc = 0.f;
#pragma unroll
      for (int k = 0; k < FE; ++k) acc += ea[k] * s_We[k * DD + d];
#pragma unroll
      for (int k = 0; k < FF; ++k) acc += ef[k] * s_Wf[k * DD + d];
      acc = acc > 0.f ? acc : 0.f;
      atomicAdd(&s_agg[dst * DD + d], acc);
    }
  }
  __syncthreads();

  // residual node update; write x-part of seq (bf16)
  const float* xg = x + (size_t)g * NN * FX;
  unsigned short* srow = seq_bf + (size_t)g * GIN;
  for (int n = tid; n < NN; n += 256) {
    float xv[FX];
#pragma unroll
    for (int k = 0; k < FX; ++k) xv[k] = xg[n * FX + k];
#pragma unroll
    for (int d = 0; d < DD; ++d) {
      float acc = s_agg[n * DD + d];
#pragma unroll
      for (int k = 0; k < FX; ++k) acc += xv[k] * s_Wx[k * DD + d];
      acc = acc > 0.f ? acc : 0.f;
      float xo = xv[d] + acc;
      s_xout[n * DD + d] = xo;
      srow[n * DD + d]   = f2bf(xo);
    }
  }
  __syncthreads();

  // residual edge update; write e-part of seq (bf16)
  for (int e = tid; e < EE; e += 256) {
    const int src = wrap80(ei[e] - mn);
    const int dst = wrap80(ei[EE + e] - mn);
    const float* ea = eag + e * FE;
    float feat[2 * DD + FE];
#pragma unroll
    for (int k = 0; k < DD; ++k) feat[k]          = s_xout[src * DD + k];
#pragma unroll
    for (int k = 0; k < DD; ++k) feat[DD + k]     = s_xout[dst * DD + k];
#pragma unroll
    for (int k = 0; k < FE; ++k) feat[2 * DD + k] = ea[k];
#pragma unroll
    for (int d = 0; d < DD; ++d) {
      float acc = 0.f;
#pragma unroll
      for (int k = 0; k < 2 * DD + FE; ++k) acc += feat[k] * s_Wedge[k * DD + d];
      acc = acc > 0.f ? acc : 0.f;
      srow[NN * DD + e * DD + d] = f2bf(ea[d] + acc);
    }
  }
}

// ---------------- GRU gates (elementwise, fp32) ----------------
__global__ void init_h_kernel(float* __restrict__ h, unsigned short* __restrict__ hbf) {
  int i = blockIdx.x * 256 + threadIdx.x;
  if (i < BB * HH) { h[i] = 0.f; hbf[i] = 0; }
}

__global__ void gru_gate_kernel(const float* __restrict__ gi, const float* __restrict__ gh,
                                const float* __restrict__ h_prev, float* __restrict__ h_new,
                                unsigned short* __restrict__ h_bf,
                                unsigned short* __restrict__ y_bf, int t) {
  int idx = blockIdx.x * 256 + threadIdx.x;
  if (idx >= BB * HH) return;
  int b = idx / HH, j = idx % HH;
  const float* gir = gi + (size_t)(b * SS + t) * G3H;
  const float* ghr = gh + (size_t)b * G3H;
  float ir = gir[j], iz = gir[HH + j], in = gir[2 * HH + j];
  float hr = ghr[j], hz = ghr[HH + j], hn = ghr[2 * HH + j];
  float r = 1.f / (1.f + __expf(-(ir + hr)));
  float z = 1.f / (1.f + __expf(-(iz + hz)));
  float n = tanhf(in + r * hn);
  float hv = (1.f - z) * n + z * h_prev[idx];
  h_new[idx] = hv;
  h_bf[idx]  = f2bf(hv);
  y_bf[(size_t)b * YW + t * HH + j] = f2bf(hv);
}

// ---------------- launch ----------------
static inline size_t align256(size_t x) { return (x + 255) & ~(size_t)255; }

extern "C" void kernel_launch(void* const* d_in, const int* in_sizes, int n_in,
                              void* d_out, int out_size, void* d_ws, size_t ws_size,
                              hipStream_t stream) {
  const float* x          = (const float*)d_in[0];
  const int*   edge_index = (const int*)  d_in[1];
  const float* edge_f     = (const float*)d_in[2];
  const float* edge_attr  = (const float*)d_in[3];
  const float* Wx         = (const float*)d_in[4];
  const float* We         = (const float*)d_in[5];
  const float* Wf         = (const float*)d_in[6];
  const float* Wedge      = (const float*)d_in[7];
  const float* w_ih       = (const float*)d_in[8];
  const float* w_hh       = (const float*)d_in[9];
  const float* b_ih       = (const float*)d_in[10];
  const float* b_hh       = (const float*)d_in[11];
  const float* mlp_w1     = (const float*)d_in[12];
  const float* mlp_b1     = (const float*)d_in[13];
  const float* mlp_w2     = (const float*)d_in[14];
  const float* mlp_b2     = (const float*)d_in[15];
  float* out = (float*)d_out;

  // workspace carve-up
  char* p = (char*)d_ws; size_t off = 0;
  auto carve = [&](size_t bytes) { void* r = p + off; off = align256(off + bytes); return r; };
  unsigned short* seq_bf  = (unsigned short*)carve((size_t)BB * SS * GIN * 2);
  unsigned short* wih_bf  = (unsigned short*)carve((size_t)G3H * GIN * 2);
  unsigned short* whh_bf  = (unsigned short*)carve((size_t)G3H * HH * 2);
  unsigned short* w1t_bf  = (unsigned short*)carve((size_t)MLP1 * YW * 2);
  unsigned short* w2t_bf  = (unsigned short*)carve((size_t)OUTF * MLP1 * 2);
  float*          gi      = (float*)carve((size_t)BB * SS * G3H * 4);
  float*          gh      = (float*)carve((size_t)BB * G3H * 4);
  float*          hA      = (float*)carve((size_t)BB * HH * 4);
  float*          hB      = (float*)carve((size_t)BB * HH * 4);
  unsigned short* h_bf    = (unsigned short*)carve((size_t)BB * HH * 2);
  unsigned short* y_bf    = (unsigned short*)carve((size_t)BB * YW * 2);
  unsigned short* hid_bf  = (unsigned short*)carve((size_t)BB * MLP1 * 2);
  (void)ws_size; (void)n_in; (void)in_sizes; (void)out_size;

  // 1) weights -> bf16, [N,K] layout
  { int n = G3H * GIN;  cvt_kernel<<<(n + 255) / 256, 256, 0, stream>>>(w_ih, wih_bf, n); }
  { int n = G3H * HH;   cvt_kernel<<<(n + 255) / 256, 256, 0, stream>>>(w_hh, whh_bf, n); }
  { int n = YW * MLP1;  cvt_t_kernel<<<(n + 255) / 256, 256, 0, stream>>>(mlp_w1, w1t_bf, YW, MLP1); }
  { int n = MLP1 * OUTF; cvt_t_kernel<<<(n + 255) / 256, 256, 0, stream>>>(mlp_w2, w2t_bf, MLP1, OUTF); }

  // 2) GNN -> seq (bf16), one block per graph
  gnn_kernel<<<BB * SS, 256, 0, stream>>>(x, edge_index, edge_f, edge_attr,
                                          Wx, We, Wf, Wedge, seq_bf);

  // 3) gi = seq @ w_ih^T + b_ih   (M=10240, N=2400, K=5376)
  {
    dim3 grid((BB * SS) / 256, (G3H + 63) / 64);
    gemm_bf16_kernel<<<grid, 256, 0, stream>>>(seq_bf, wih_bf, b_ih, gi, nullptr,
                                               BB * SS, G3H, GIN);
  }

  // 4) GRU recurrence over S=5 steps
  init_h_kernel<<<(BB * HH + 255) / 256, 256, 0, stream>>>(hA, h_bf);
  float* hp = hA; float* hn = hB;
  for (int t = 0; t < SS; ++t) {
    dim3 grid(BB / 256, (G3H + 63) / 64);
    gemm_bf16_kernel<<<grid, 256, 0, stream>>>(h_bf, whh_bf, b_hh, gh, nullptr,
                                               BB, G3H, HH);
    gru_gate_kernel<<<(BB * HH + 255) / 256, 256, 0, stream>>>(gi, gh, hp, hn, h_bf, y_bf, t);
    float* tmp = hp; hp = hn; hn = tmp;
  }

  // 5) MLP layer 1: hid = y @ w1 + b1  (bf16 out fused)   M=2048, N=1280, K=4000
  {
    dim3 grid(BB / 256, MLP1 / 64);
    gemm_bf16_kernel<<<grid, 256, 0, stream>>>(y_bf, w1t_bf, mlp_b1, nullptr, hid_bf,
                                               BB, MLP1, YW);
  }
  // 6) MLP layer 2: out = hid @ w2 + b2  (fp32 to d_out)  M=2048, N=400, K=1280
  {
    dim3 grid(BB / 256, (OUTF + 63) / 64);
    gemm_bf16_kernel<<<grid, 256, 0, stream>>>(hid_bf, w2t_bf, mlp_b2, out, nullptr,
                                               BB, OUTF, MLP1);
  }
}
